// BahdanauMultiheadAttention_9637906612352
// MI455X (gfx1250) — matrix operand order repaired and checked by reading, not compiled, not run on hardware
//
#include <hip/hip_runtime.h>
#include <hip/hip_bf16.h>

// ---------------------------------------------------------------------------
// Problem constants (B=4, L=256, D=512, H=8, DK=64)
// ---------------------------------------------------------------------------
#define BB 4
#define LL 256
#define DD 512
#define HH 8
#define DKK 64

typedef __attribute__((ext_vector_type(16))) __bf16 v16bf;
typedef __attribute__((ext_vector_type(8)))  float  v8f;

union Frag16 {
    int   i[8];   // 8 dwords = 16 packed bf16
    v16bf v;
};

__device__ __forceinline__ unsigned int f2bf(float f) {
    // round-half-up fp32 -> bf16 (result in low 16 bits); 2 VALU ops
    return (__float_as_uint(f) + 0x8000u) >> 16;
}

__device__ __forceinline__ unsigned int pack2bf(float lo, float hi) {
    // 2x v_add_u32 + 1x v_perm_b32: splice high halves of the two rounded
    // fp32 values into one packed bf16 dword (lo -> [15:0], hi -> [31:16]).
    const unsigned int ul = __float_as_uint(lo) + 0x8000u;
    const unsigned int uh = __float_as_uint(hi) + 0x8000u;
    return __builtin_amdgcn_perm(uh, ul, 0x07060302u);
}

__device__ __forceinline__ float fast_tanh(float x) {
#if __has_builtin(__builtin_amdgcn_tanhf)
    return __builtin_amdgcn_tanhf(x);   // v_tanh_f32 on gfx1250
#else
    return tanhf(x);
#endif
}

// ---------------------------------------------------------------------------
// Kernel 1/4: C[M=1024, N=512] = A[1024,512] @ W[512,512] + bias
// fp32 inputs converted to bf16 in LDS; v_wmma_f32_16x16x32_bf16.
// SCATTER==1 : write into (B,H,L,DK) layout (row m=b*L+l, col n=h*64+dk)
// SCATTER==0 : plain row-major
// Block: 256 threads = 8 waves, arranged 2 (M) x 4 (N); wave tile 32x32.
// Block tile 64 x 128; K step 32.
// ---------------------------------------------------------------------------
template <int SCATTER>
__global__ __launch_bounds__(256) void gemm_bias_kernel(
    const float* __restrict__ A, const float* __restrict__ W,
    const float* __restrict__ bias, float* __restrict__ dst)
{
    __shared__ alignas(16) unsigned short As[64][32];   // [M][K] bf16
    __shared__ alignas(16) unsigned short Bs[128][32];  // [N][K] bf16

    const int tid  = threadIdx.x;
    const int lane = tid & 31;
    const int wave = tid >> 5;
    const int waveM = wave & 1;   // 0..1 -> 32-row group
    const int waveN = wave >> 1;  // 0..3 -> 32-col group
    const int m0 = blockIdx.x * 64;
    const int n0 = blockIdx.y * 128;

    const v8f vzero = {0.f, 0.f, 0.f, 0.f, 0.f, 0.f, 0.f, 0.f};
    v8f acc[2][2];
    acc[0][0] = vzero; acc[0][1] = vzero; acc[1][0] = vzero; acc[1][1] = vzero;

    const int lr    = lane & 15;
    const int koffA = (lane & 16) ? 16 : 0;   // A: K contiguous {0-15} vs {16-31}
    const int koffB = (lane & 16) ? 16 : 0;   // B: K contiguous {0-15} vs {16-31}

    // staging thread mappings
    const int aRow = tid >> 2;               // 0..63
    const int aKc  = (tid & 3) * 8;          // 0,8,16,24
    const int bK   = (tid & 15) * 2;         // 0..30 (K pair)
    const int bNc  = (tid >> 4) * 8;         // 0..120

    for (int kb = 0; kb < DD; kb += 32) {
        // --- stage A tile 64x32 (fp32 -> bf16, packed dword stores) ---
        {
            const float* ap = A + (m0 + aRow) * DD + kb + aKc;
            if (kb + 32 < DD) __builtin_prefetch(ap + 32, 0, 3);
            unsigned int* as = (unsigned int*)&As[aRow][aKc];
#pragma unroll
            for (int j = 0; j < 4; ++j) as[j] = pack2bf(ap[2 * j], ap[2 * j + 1]);
        }
        // --- stage W tile 32x128, transposed to [N][K]; thread owns a K-pair ---
        {
            const float* wp0 = W + (kb + bK) * DD + n0 + bNc;
            const float* wp1 = wp0 + DD;
            if (kb + 32 < DD) __builtin_prefetch(wp0 + 32 * DD, 0, 3);
#pragma unroll
            for (int j = 0; j < 8; ++j)
                *(unsigned int*)&Bs[bNc + j][bK] = pack2bf(wp0[j], wp1[j]);
        }
        __syncthreads();

        Frag16 aF[2], bF[2];
#pragma unroll
        for (int s = 0; s < 2; ++s) {
            const int row = waveM * 32 + s * 16 + lr;
#pragma unroll
            for (int j = 0; j < 8; ++j) {
                const int k = koffA + 2 * j;
                aF[s].i[j] = *(const int*)&As[row][k];
            }
        }
#pragma unroll
        for (int t = 0; t < 2; ++t) {
            const int col = waveN * 32 + t * 16 + lr;
#pragma unroll
            for (int j = 0; j < 8; ++j)
                bF[t].i[j] = *(const int*)&Bs[col][koffB + 2 * j];
        }
#pragma unroll
        for (int s = 0; s < 2; ++s)
#pragma unroll
            for (int t = 0; t < 2; ++t)
                acc[s][t] = __builtin_amdgcn_wmma_f32_16x16x32_bf16(
                    false, aF[s].v, false, bF[t].v, (short)0, acc[s][t],
                    false, false);
        __syncthreads();
    }

    // --- epilogue: bias + store (C/D layout: vgpr v -> row v or v+8) ---
#pragma unroll
    for (int t = 0; t < 2; ++t) {
        const int n = n0 + waveN * 32 + t * 16 + (lane & 15);
        const float bv = bias[n];
#pragma unroll
        for (int s = 0; s < 2; ++s) {
#pragma unroll
            for (int vv = 0; vv < 8; ++vv) {
                const int m = m0 + waveM * 32 + s * 16 + ((lane < 16) ? vv : vv + 8);
                const float val = acc[s][t][vv] + bv;
                if (SCATTER) {
                    const int b = m >> 8, l = m & 255;
                    const int h = n >> 6, dk = n & 63;
                    dst[(((b * HH) + h) * LL + l) * DKK + dk] = val;
                } else {
                    dst[m * DD + n] = val;
                }
            }
        }
    }
}

// ---------------------------------------------------------------------------
// Kernel 2: Bahdanau scores + softmax.
// One block per (b, h, query i); lane t owns key j=t.
// s_j = sum_d vp[h,d] * tanh(qh[b,h,i,d] + kh[b,h,j,d]); mask; softmax.
// Emits bf16 probabilities for the WMMA AV GEMM.
// ---------------------------------------------------------------------------
__global__ __launch_bounds__(256) void score_softmax_kernel(
    const float* __restrict__ qh, const float* __restrict__ kh,
    const float* __restrict__ vp, const int* __restrict__ mask,
    unsigned short* __restrict__ attn)
{
    const int i = blockIdx.x;
    const int h = blockIdx.y;
    const int b = blockIdx.z;
    const int t = threadIdx.x;        // key index j
    const int bh = b * HH + h;

    __shared__ float qs[DKK];
    __shared__ float vps[DKK];
    __shared__ float red[256];

    if (t < DKK)            qs[t]        = qh[(bh * LL + i) * DKK + t];
    else if (t < 2 * DKK)   vps[t - DKK] = vp[h * DKK + (t - DKK)];
    __syncthreads();

    const float4* k4 = (const float4*)(kh + (bh * LL + t) * DKK);
    float s = 0.f;
#pragma unroll
    for (int d4 = 0; d4 < DKK / 4; ++d4) {
        const float4 kv = k4[d4];
        s += vps[4 * d4 + 0] * fast_tanh(qs[4 * d4 + 0] + kv.x);
        s += vps[4 * d4 + 1] * fast_tanh(qs[4 * d4 + 1] + kv.y);
        s += vps[4 * d4 + 2] * fast_tanh(qs[4 * d4 + 2] + kv.z);
        s += vps[4 * d4 + 3] * fast_tanh(qs[4 * d4 + 3] + kv.w);
    }

    if (mask[(b * LL + i) * LL + t] == 0) s = -1.0e9f;

    // block softmax over 256 keys
    red[t] = s;
    __syncthreads();
    for (int off = 128; off > 0; off >>= 1) {
        if (t < off) red[t] = fmaxf(red[t], red[t + off]);
        __syncthreads();
    }
    const float mx = red[0];
    __syncthreads();
    const float e = __expf(s - mx);
    red[t] = e;
    __syncthreads();
    for (int off = 128; off > 0; off >>= 1) {
        if (t < off) red[t] += red[t + off];
        __syncthreads();
    }
    const float inv = 1.f / red[0];

    attn[(bh * LL + i) * LL + t] = (unsigned short)f2bf(e * inv);
}

// ---------------------------------------------------------------------------
// Kernel 3: per-(b,h) AV GEMM: ctx[256,64] = attn[256,256](bf16) @ vh[256,64]
// The attn tile is a pure byte copy -> use gfx1250 async global->LDS DMA
// (GLOBAL_LOAD_ASYNC_TO_LDS_B128, ASYNCcnt-tracked, no VGPR round-trip).
// Block: 256 threads = 8 waves arranged 4 (M) x 2 (N); wave tile 16x32.
// Block tile 64 x 64; K step 32. Output written directly in (B,L,D) layout.
// ---------------------------------------------------------------------------
__global__ __launch_bounds__(256) void av_gemm_kernel(
    const unsigned short* __restrict__ attn, const float* __restrict__ vh,
    float* __restrict__ ctx)
{
    __shared__ alignas(16) unsigned short As[64][32];  // attn tile [M][K]
    __shared__ alignas(16) unsigned short Bs[64][32];  // vh tile [N][K]

    const int tid  = threadIdx.x;
    const int lane = tid & 31;
    const int wave = tid >> 5;
    const int waveM = wave & 3;   // 0..3 -> 16-row group
    const int waveN = wave >> 2;  // 0..1 -> 32-col group
    const int m0 = blockIdx.x * 64;
    const int bh = blockIdx.y;
    const int b  = bh >> 3;
    const int h  = bh & 7;

    const unsigned short* aBase = attn + bh * (LL * LL);
    const float*          vBase = vh   + bh * (LL * DKK);

    const v8f vzero = {0.f, 0.f, 0.f, 0.f, 0.f, 0.f, 0.f, 0.f};
    v8f acc[2];
    acc[0] = vzero; acc[1] = vzero;

    const int lr    = lane & 15;
    const int koffA = (lane & 16) ? 16 : 0;
    const int koffB = (lane & 16) ? 16 : 0;

    // staging thread mappings
    const int aRow = tid >> 2;               // 0..63
    const int aKc  = (tid & 3) * 8;          // 0,8,16,24  (16B per thread)
    const int bK   = (tid & 15) * 2;         // 0..30 (K pair)
    const int bNc  = (tid >> 4) * 4;         // 0..60

    const unsigned int ldsA =
        (unsigned int)(unsigned long long)&As[aRow][aKc];  // flat->DS addr (addr[31:0])

    for (int kb = 0; kb < LL; kb += 32) {
        // --- attn tile 64x32 bf16: async DMA global -> LDS (16B/lane) ---
        {
            const unsigned short* gsrc = aBase + (m0 + aRow) * LL + kb + aKc;
            asm volatile("global_load_async_to_lds_b128 %0, %1, off"
                         :: "v"(ldsA), "v"(gsrc) : "memory");
        }
        // --- vh tile 32x64 fp32 -> [N][K] bf16; thread owns a K-pair ---
        {
            const float* vp0 = vBase + (kb + bK) * DKK + bNc;
            const float* vp1 = vp0 + DKK;
#pragma unroll
            for (int j = 0; j < 4; ++j)
                *(unsigned int*)&Bs[bNc + j][bK] = pack2bf(vp0[j], vp1[j]);
        }
        asm volatile("s_wait_asynccnt 0x0" ::: "memory");
        __syncthreads();

        Frag16 aF, bF[2];
        {
            const int row = waveM * 16 + lr;
#pragma unroll
            for (int j = 0; j < 8; ++j) {
                const int k = koffA + 2 * j;
                aF.i[j] = *(const int*)&As[row][k];
            }
        }
#pragma unroll
        for (int t = 0; t < 2; ++t) {
            const int col = waveN * 32 + t * 16 + lr;
#pragma unroll
            for (int j = 0; j < 8; ++j)
                bF[t].i[j] = *(const int*)&Bs[col][koffB + 2 * j];
        }
#pragma unroll
        for (int t = 0; t < 2; ++t)
            acc[t] = __builtin_amdgcn_wmma_f32_16x16x32_bf16(
                false, aF.v, false, bF[t].v, (short)0, acc[t], false, false);
        __syncthreads();
    }

#pragma unroll
    for (int t = 0; t < 2; ++t) {
#pragma unroll
        for (int vv = 0; vv < 8; ++vv) {
            const int l = m0 + waveM * 16 + ((lane < 16) ? vv : vv + 8);
            const int n = waveN * 32 + t * 16 + (lane & 15);
            ctx[(b * LL + l) * DD + h * DKK + n] = acc[t][vv];
        }
    }
}

// ---------------------------------------------------------------------------
// Launch. d_in order: q,k,v,mask,Wq,bq,Wk,bk,Wv,bv,vp,W0,b0
// Workspace: qh | kh | vh | ctx (fp32, 2MB each) | attn (bf16, 4MB) = 12MB
// ---------------------------------------------------------------------------
extern "C" void kernel_launch(void* const* d_in, const int* in_sizes, int n_in,
                              void* d_out, int out_size, void* d_ws, size_t ws_size,
                              hipStream_t stream) {
    const float* q    = (const float*)d_in[0];
    const float* k    = (const float*)d_in[1];
    const float* v    = (const float*)d_in[2];
    const int*   mask = (const int*)  d_in[3];
    const float* Wq   = (const float*)d_in[4];
    const float* bq   = (const float*)d_in[5];
    const float* Wk   = (const float*)d_in[6];
    const float* bk   = (const float*)d_in[7];
    const float* Wv   = (const float*)d_in[8];
    const float* bv   = (const float*)d_in[9];
    const float* vp   = (const float*)d_in[10];
    const float* W0   = (const float*)d_in[11];
    const float* b0   = (const float*)d_in[12];

    float* ws   = (float*)d_ws;
    const size_t nBHLD = (size_t)BB * HH * LL * DKK;  // 524288
    float* qh  = ws;
    float* kh  = ws + nBHLD;
    float* vh  = ws + 2 * nBHLD;
    float* ctx = ws + 3 * nBHLD;
    unsigned short* attn = (unsigned short*)(ws + 4 * nBHLD);

    const dim3 blk(256);
    const dim3 gGemm(BB * LL / 64, DD / 128);           // (16, 4)

    // Q/K/V projections -> (B,H,L,DK)
    gemm_bias_kernel<1><<<gGemm, blk, 0, stream>>>(q, Wq, bq, qh);
    gemm_bias_kernel<1><<<gGemm, blk, 0, stream>>>(k, Wk, bk, kh);
    gemm_bias_kernel<1><<<gGemm, blk, 0, stream>>>(v, Wv, bv, vh);

    // Bahdanau scores + softmax -> bf16 attn
    const dim3 gScore(LL, HH, BB);                      // (256, 8, 4)
    score_softmax_kernel<<<gScore, blk, 0, stream>>>(qh, kh, vp, mask, attn);

    // attn @ vh -> ctx in (B,L,D) layout
    const dim3 gAV(LL / 64, BB * HH);                   // (4, 32)
    av_gemm_kernel<<<gAV, blk, 0, stream>>>(attn, vh, ctx);

    // ctx @ W0 + b0 -> output
    gemm_bias_kernel<0><<<gGemm, blk, 0, stream>>>(ctx, W0, b0, (float*)d_out);
}
